// VQVAE_81535659147903
// MI455X (gfx1250) — matrix operand order repaired and checked by reading, not compiled, not run on hardware
//
#include <hip/hip_runtime.h>
#include <hip/hip_bf16.h>
#include <math.h>

typedef _Float16 v16h __attribute__((ext_vector_type(16)));
typedef float    v8f  __attribute__((ext_vector_type(8)));
typedef unsigned int u32x4 __attribute__((ext_vector_type(4)));
typedef int          i32x8 __attribute__((ext_vector_type(8)));
typedef int          i32x4 __attribute__((ext_vector_type(4)));

#define BETA 0.1f
#define EPSV 1e-10f

// LDS layout (single static block -> offset 0):
//   A buffers: 2 x 16896 B  (fp32 128x33 dwords for TDM mode, fp16 128x34 halves for gather modes)
//   B buffers: 2 x 4352 B   (fp16 64x34 halves, TDM-padded: 16 dwords + 1 pad dword per row)
#define SA  16896u
#define BOFF 33792u
#define SB  4352u
#define SMEM_BYTES 42496

struct GemmP {
  const float*    X;     // A source (fp32)
  const _Float16* W;     // packed B, N x K row-major fp16 (n-major)
  const float*    bias;
  const float*    resid;
  const float*    zn;
  const float*    cn;
  float*          Y;
  int M, N, Ktot;
  int Bn, Cin, Hin, Win, Hout, Wout, KH, KW, stride, pad, relu;
};

// 2-D TDM tile load: tile (tile1 rows x tile0 elems), row stride `stride` elems,
// OOB (x>=td0 or y>=td1) returns zeros. dsz: 0=1B,1=2B,2=4B. padi/pada: LDS padding codes.
__device__ __forceinline__ void tdm_load_2d(unsigned lds_off, const void* gaddr,
    unsigned dsz, unsigned tile0, unsigned tile1, unsigned td0, unsigned td1,
    unsigned long stride, unsigned padi, unsigned pada) {
  unsigned long ga = (unsigned long)gaddr;
  u32x4 g0;
  g0[0] = 1u;                                   // count=1 (valid descriptor)
  g0[1] = lds_off;                              // lds_addr
  g0[2] = (unsigned)(ga & 0xffffffffu);         // global_addr lo
  g0[3] = (unsigned)((ga >> 32) & 0x01ffffffu) | 0x80000000u;  // addr hi | type=2
  i32x8 g1;
  g1[0] = (int)((dsz << 16) | (1u << 20) | (padi << 22) | (pada << 25)); // pad_enable=1
  g1[1] = (int)((td0 & 0xffffu) << 16);                  // tensor_dim0 lo16 @bits63:48
  g1[2] = (int)((td0 >> 16) | ((td1 & 0xffffu) << 16));  // td0 hi | td1 lo
  g1[3] = (int)((td1 >> 16) | (tile0 << 16));            // td1 hi | tile_dim0
  g1[4] = (int)(tile1 & 0xffffu);                        // tile_dim1 (tile_dim2=0)
  g1[5] = (int)(stride & 0xffffffffu);                   // tensor_dim0_stride lo
  g1[6] = (int)((stride >> 32) & 0xffffu);               // stride hi
  g1[7] = 0;
  i32x4 zz; zz[0] = 0; zz[1] = 0; zz[2] = 0; zz[3] = 0;
  i32x8 z8; z8[0] = 0; z8[1] = 0; z8[2] = 0; z8[3] = 0;
  z8[4] = 0; z8[5] = 0; z8[6] = 0; z8[7] = 0;
  __builtin_amdgcn_tensor_load_to_lds(g0, g1, zz, zz, z8, 0);
}

// GATHER: 0=conv im2col, 1=convT gather, 2=NCHW channel gather, 3=dense row-major (TDM)
// EPI:    0=bias(+resid)(+relu)->NCHW, 1=zn+cn-2*acc->rowmajor, 2=plain rowmajor
template<int GATHER, int EPI>
__global__ __launch_bounds__(256)
void gemm_wmma(GemmP p) {
  __shared__ __attribute__((aligned(16))) char smem[SMEM_BYTES];
  int t     = threadIdx.x;
  int mBase = blockIdx.x * 128;
  int nBase = blockIdx.y * 64;
  int wv = t >> 5, lane = t & 31;
  int hl = lane >> 4, mr = lane & 15;

  v8f acc[4] = {};

  // ---- per-thread A-row decode (hoisted out of K loop) ----
  int kb   = (t & 1) * 16;
  int arow = t >> 1;
  int am   = mBase + arow; if (am >= p.M) am = p.M - 1;
  int hw   = p.Hout * p.Wout;
  int ab   = am / hw; int arem = am - ab * hw;
  int aoh  = arem / p.Wout; int aow = arem - aoh * p.Wout;
  size_t hwim = (size_t)p.Hin * p.Win;
  const float* Xb;
  if (GATHER == 2) Xb = p.X + (size_t)ab * p.Ktot * hwim + (size_t)aoh * p.Win + aow;
  else             Xb = p.X + (size_t)ab * p.Cin * hwim;
  int ihb = aoh * p.stride - p.pad, iwb = aow * p.stride - p.pad;

  float regs[16];
  auto loadA = [&](int k0) {   // global -> regs (gather modes)
    if (GATHER == 3) return;
    int ks0 = k0 + kb;
    if (GATHER == 2) {
      #pragma unroll
      for (int i = 0; i < 16; ++i) {
        int k = ks0 + i;
        regs[i] = (k < p.Ktot) ? Xb[(size_t)k * hwim] : 0.f;
      }
    } else {
      int kk2 = p.KH * p.KW;
      int ci = ks0 / kk2; int rem = ks0 - ci * kk2;
      int kr = rem / p.KW; int ks = rem - kr * p.KW;
      #pragma unroll
      for (int i = 0; i < 16; ++i) {
        float v = 0.f;
        if (ks0 + i < p.Ktot) {
          if (GATHER == 0) {
            int ih = ihb + kr, iw = iwb + ks;
            if (ih >= 0 && ih < p.Hin && iw >= 0 && iw < p.Win)
              v = Xb[(size_t)ci * hwim + ih * p.Win + iw];
          } else {
            int t1 = aoh + p.pad - kr, t2 = aow + p.pad - ks;
            if (!(((t1 | t2) < 0) || (t1 & 1) || (t2 & 1))) {
              int ih = t1 >> 1, iw = t2 >> 1;
              if (ih < p.Hin && iw < p.Win)
                v = Xb[(size_t)ci * hwim + ih * p.Win + iw];
            }
          }
        }
        regs[i] = v;
        if (++ks == p.KW) { ks = 0; if (++kr == p.KH) { kr = 0; ++ci; } }
      }
    }
  };
  auto storeA = [&](int buf) {  // regs -> LDS fp16
    if (GATHER == 3) return;
    _Float16* Ah = (_Float16*)(smem + buf * SA);
    #pragma unroll
    for (int i = 0; i < 16; ++i) Ah[arow * 34 + kb + i] = (_Float16)regs[i];
  };
  auto tdmIssue = [&](int k0, int buf) {  // wave 0; TDM issue is EXEC-independent
    if (t == 0) {
      unsigned remk = (unsigned)(p.Ktot - k0);
      if (GATHER == 3)
        tdm_load_2d(buf * SA, p.X + (size_t)mBase * p.Ktot + k0, 2u,
                    32u, 128u, remk, 128u, (unsigned long)p.Ktot, 4u, 0u);
      tdm_load_2d(BOFF + buf * SB, p.W + (size_t)nBase * p.Ktot + k0, 1u,
                  32u, 64u, remk, (unsigned)(p.N - nBase),
                  (unsigned long)p.Ktot, 3u, 0u);
    }
  };
  auto compute = [&](int cur) {
    v16h a;
    int row = wv * 16 + mr;
    if (GATHER == 3) {
      const float* Af = (const float*)(smem + cur * SA);
      #pragma unroll
      for (int v = 0; v < 8; ++v) {
        int kk = 2 * v + ((v < 4) ? 0 : 8) + hl * 8;
        a[2 * v]     = (_Float16)Af[row * 33 + kk];
        a[2 * v + 1] = (_Float16)Af[row * 33 + kk + 1];
      }
    } else {
      const _Float16* Ah = (const _Float16*)(smem + cur * SA);
      #pragma unroll
      for (int v = 0; v < 8; ++v) {
        int kk = 2 * v + ((v < 4) ? 0 : 8) + hl * 8;
        a[2 * v]     = Ah[row * 34 + kk];
        a[2 * v + 1] = Ah[row * 34 + kk + 1];
      }
    }
    const _Float16* Bh = (const _Float16*)(smem + BOFF + cur * SB);
    #pragma unroll
    for (int c = 0; c < 4; ++c) {
      v16h b;
      #pragma unroll
      for (int v = 0; v < 8; ++v) {
        int kk = 2 * v + ((v < 4) ? 0 : 8) + hl * 8;
        b[2 * v]     = Bh[(c * 16 + mr) * 34 + kk];
        b[2 * v + 1] = Bh[(c * 16 + mr) * 34 + kk + 1];
      }
      acc[c] = __builtin_amdgcn_wmma_f32_16x16x32_f16(false, a, false, b,
                                                      (short)0, acc[c], false, false);
    }
  };

  // ---- software-pipelined K loop (double-buffered LDS, TDM async B/A) ----
  tdmIssue(0, 0);
  loadA(0);
  storeA(0);
  __builtin_amdgcn_s_wait_tensorcnt(0);
  __syncthreads();
  int steps = (p.Ktot + 31) >> 5;
  for (int s = 0; s < steps; ++s) {
    int cur = s & 1, nxt = cur ^ 1;
    int k0n = (s + 1) << 5;
    bool more = k0n < p.Ktot;
    if (more) { tdmIssue(k0n, nxt); loadA(k0n); }
    compute(cur);
    if (more) storeA(nxt);
    __builtin_amdgcn_s_wait_tensorcnt(0);
    __syncthreads();
  }

  // ---- epilogue (decode hoisted, incremental carries) ----
  int n0 = nBase + (lane & 15);
  if (EPI == 0) {
    int m0r = mBase + wv * 16 + 8 * hl;
    int b0 = m0r / hw; int rr = m0r - b0 * hw;
    int oh0 = rr / p.Wout; int ow0 = rr - oh0 * p.Wout;
    #pragma unroll
    for (int r = 0; r < 8; ++r) {
      size_t base = (size_t)b0 * p.N * hw + (size_t)oh0 * p.Wout + ow0;
      #pragma unroll
      for (int c = 0; c < 4; ++c) {
        int n = n0 + c * 16;
        if (n < p.N) {
          float v = acc[c][r];
          if (p.bias)  v += p.bias[n];
          size_t oi = base + (size_t)n * hw;
          if (p.resid) v += p.resid[oi];
          if (p.relu)  v = fmaxf(v, 0.f);
          p.Y[oi] = v;
        }
      }
      if (++ow0 == p.Wout) { ow0 = 0; if (++oh0 == p.Hout) { oh0 = 0; ++b0; } }
    }
  } else {
    #pragma unroll
    for (int c = 0; c < 4; ++c) {
      int n = n0 + c * 16;
      if (n >= p.N) continue;
      #pragma unroll
      for (int r = 0; r < 8; ++r) {
        int m = mBase + wv * 16 + r + 8 * hl;
        if (m >= p.M) continue;
        float v = acc[c][r];
        if (EPI == 1) p.Y[(size_t)m * p.N + n] = p.zn[m] + p.cn[n] - 2.f * v;
        else          p.Y[(size_t)m * p.N + n] = v;
      }
    }
  }
}

// ---------------- weight packing (n-major: o[n*K + k]) ----------------
__global__ void pack_conv(const float* w, _Float16* o, int Cout, int Cin, int KH, int KW) {
  int i = blockIdx.x * 256 + threadIdx.x;
  int K = Cin * KH * KW; int tot = K * Cout;
  if (i >= tot) return;
  int n = i / K, k = i - n * K;
  int kk = KH * KW; int ci = k / kk; int r = (k - ci * kk) / KW; int s = k % KW;
  o[(size_t)n * K + k] = (_Float16)w[(((size_t)n * Cin + ci) * KH + r) * KW + s];
}
__global__ void pack_convT(const float* w, _Float16* o, int Cout, int Cin, int KH, int KW) {
  int i = blockIdx.x * 256 + threadIdx.x;
  int K = Cin * KH * KW; int tot = K * Cout;
  if (i >= tot) return;
  int n = i / K, k = i - n * K;
  int kk = KH * KW; int ci = k / kk; int r = (k - ci * kk) / KW; int s = k % KW;
  o[(size_t)n * K + k] = (_Float16)w[(((size_t)ci * Cout + n) * KH + r) * KW + s];
}
__global__ void pack_cbT(const float* cb, _Float16* o) {  // -> [n=d(256)][k=code(1024)]
  int i = blockIdx.x * 256 + threadIdx.x;
  if (i >= 262144) return;
  int n = i >> 10, kc = i & 1023;
  o[(size_t)n * 1024 + kc] = (_Float16)cb[(size_t)kc * 256 + n];
}
__global__ void pack_cast(const float* s, _Float16* o, int n) {
  int i = blockIdx.x * 256 + threadIdx.x;
  if (i < n) o[i] = (_Float16)s[i];
}

// ---------------- VQ / BA kernels ----------------
__global__ void row_norm_nchw(const float* z, float* zn) {
  int m = blockIdx.x * 256 + threadIdx.x;
  if (m >= 32768) return;
  int b = m >> 8, hw = m & 255, h = hw >> 4, w = hw & 15;
  float s = 0.f;
  for (int d = 0; d < 256; ++d) {
    float v = z[(((size_t)b * 256 + d) * 16 + h) * 16 + w];
    s += v * v;
  }
  zn[m] = s;
}
__global__ void cb_norm(const float* cb, float* cn) {
  int n = blockIdx.x * 256 + threadIdx.x;
  if (n >= 1024) return;
  float s = 0.f;
  for (int d = 0; d < 256; ++d) { float v = cb[(size_t)n * 256 + d]; s += v * v; }
  cn[n] = s;
}
__global__ void init_q(const float* pi, float* Q, float* lQ, float* Qs) {
  int n = blockIdx.x * 256 + threadIdx.x;
  if (n >= 1024) return;
  float q = pi[n];
  Q[n] = q; lQ[n] = logf(fmaxf(q, EPSV)); Qs[n] = 0.f;
}
__global__ void ba_softmax(const float* d, const float* lQ, float* P) {
  __shared__ float red[256];
  int m = blockIdx.x, t = threadIdx.x;
  const float* dr = d + (size_t)m * 1024;
  float l[4], e[4];
  float mx = -3.0e38f;
  #pragma unroll
  for (int j = 0; j < 4; ++j) {
    int n = t + j * 256;
    l[j] = lQ[n] - BETA * dr[n];
    mx = fmaxf(mx, l[j]);
  }
  red[t] = mx; __syncthreads();
  for (int s = 128; s > 0; s >>= 1) { if (t < s) red[t] = fmaxf(red[t], red[t + s]); __syncthreads(); }
  mx = red[0]; __syncthreads();
  float sm = 0.f;
  #pragma unroll
  for (int j = 0; j < 4; ++j) { e[j] = expf(l[j] - mx); sm += e[j]; }
  red[t] = sm; __syncthreads();
  for (int s = 128; s > 0; s >>= 1) { if (t < s) red[t] += red[t + s]; __syncthreads(); }
  float inv = 1.f / red[0];
  #pragma unroll
  for (int j = 0; j < 4; ++j) P[(size_t)m * 1024 + t + j * 256] = e[j] * inv;
}
__global__ void colsum(const float* P, float* Qs) {
  int n  = blockIdx.x * 256 + threadIdx.x;
  int r0 = blockIdx.y * 128;
  float s = 0.f;
  for (int m = r0; m < r0 + 128; ++m) s += P[(size_t)m * 1024 + n];
  atomicAdd(&Qs[n], s);
}
__global__ void update_q(float* Q, float* lQ, float* Qs) {
  int n = blockIdx.x * 256 + threadIdx.x;
  if (n >= 1024) return;
  float q = fmaxf(Qs[n] * (1.0f / 32768.0f), EPSV);
  Q[n] = q; lQ[n] = logf(q); Qs[n] = 0.f;
}
__global__ void row_argmax(const float* P, int* idx) {
  __shared__ float rv[256]; __shared__ int ri[256];
  int m = blockIdx.x, t = threadIdx.x;
  float best = -3.0e38f; int bi = 0;
  #pragma unroll
  for (int j = 0; j < 4; ++j) {
    int n = t + j * 256;
    float v = P[(size_t)m * 1024 + n];
    if (v > best) { best = v; bi = n; }
  }
  rv[t] = best; ri[t] = bi; __syncthreads();
  for (int s = 128; s > 0; s >>= 1) {
    if (t < s) {
      if (rv[t + s] > rv[t] || (rv[t + s] == rv[t] && ri[t + s] < ri[t])) {
        rv[t] = rv[t + s]; ri[t] = ri[t + s];
      }
    }
    __syncthreads();
  }
  if (t == 0) idx[m] = ri[0];
}
__global__ void scatter_zq(const float* cb, const int* idx, float* zq) {
  size_t i = (size_t)blockIdx.x * 256 + threadIdx.x;
  if (i >= 8388608) return;
  int m = (int)(i >> 8), d = (int)(i & 255);
  int b = m >> 8, hw = m & 255, h = hw >> 4, w = hw & 15;
  zq[(((size_t)b * 256 + d) * 16 + h) * 16 + w] = cb[(size_t)idx[m] * 256 + d];
}
__global__ void zero_scal(float* s) { if (threadIdx.x < 16) s[threadIdx.x] = 0.f; }
__global__ void commit_reduce(const float* z, const float* zqs, float* scal) {
  __shared__ float red[256];
  float s = 0.f;
  for (size_t i = (size_t)blockIdx.x * 256 + threadIdx.x; i < 8388608;
       i += (size_t)gridDim.x * 256) {
    int m = (int)(i >> 8), d = (int)(i & 255);
    int b = m >> 8, hw = m & 255, h = hw >> 4, w = hw & 15;
    float zf = z[(((size_t)b * 256 + d) * 16 + h) * 16 + w];
    float df = zf - zqs[i];
    s += df * df;
  }
  red[threadIdx.x] = s; __syncthreads();
  for (int st = 128; st > 0; st >>= 1) {
    if (threadIdx.x < st) red[threadIdx.x] += red[threadIdx.x + st];
    __syncthreads();
  }
  if (threadIdx.x == 0) atomicAdd(&scal[0], red[0]);
}
__global__ void finalize(const float* scal, const float* Q, const float* lQ, float* out) {
  __shared__ float red[256];
  int t = threadIdx.x;
  float s = 0.f;
  for (int n = t; n < 1024; n += 256) s += Q[n] * lQ[n];
  red[t] = s; __syncthreads();
  for (int st = 128; st > 0; st >>= 1) { if (t < st) red[t] += red[t + st]; __syncthreads(); }
  if (t == 0) {
    float mse = scal[0] / 8388608.0f;
    out[0] = 0.25f * mse;
    out[1] = mse;
    out[2] = 0.01f * red[0];
  }
}

// ---------------- host orchestration ----------------
extern "C" void kernel_launch(void* const* d_in, const int* in_sizes, int n_in,
                              void* d_out, int out_size, void* d_ws, size_t ws_size,
                              hipStream_t stream) {
  (void)in_sizes; (void)n_in; (void)out_size; (void)ws_size;
  const float* x     = (const float*)d_in[0];
  const float* ew1   = (const float*)d_in[1];
  const float* eb1   = (const float*)d_in[2];
  const float* ew2   = (const float*)d_in[3];
  const float* eb2   = (const float*)d_in[4];
  const float* ew3   = (const float*)d_in[5];
  const float* eb3   = (const float*)d_in[6];
  const float* er1aw = (const float*)d_in[7];
  const float* er1ab = (const float*)d_in[8];
  const float* er1bw = (const float*)d_in[9];
  const float* er1bb = (const float*)d_in[10];
  const float* er2aw = (const float*)d_in[11];
  const float* er2ab = (const float*)d_in[12];
  const float* er2bw = (const float*)d_in[13];
  const float* er2bb = (const float*)d_in[14];
  const float* cb    = (const float*)d_in[15];
  const float* pi    = (const float*)d_in[16];
  const float* dr1aw = (const float*)d_in[17];
  const float* dr1ab = (const float*)d_in[18];
  const float* dr1bw = (const float*)d_in[19];
  const float* dr1bb = (const float*)d_in[20];
  const float* dr2aw = (const float*)d_in[21];
  const float* dr2ab = (const float*)d_in[22];
  const float* dr2bw = (const float*)d_in[23];
  const float* dr2bb = (const float*)d_in[24];
  const float* dw1   = (const float*)d_in[25];
  const float* db1   = (const float*)d_in[26];
  const float* dwt2  = (const float*)d_in[27];
  const float* dbt2  = (const float*)d_in[28];
  const float* dwt3  = (const float*)d_in[29];
  const float* dbt3  = (const float*)d_in[30];
  float* out = (float*)d_out;

  float* F    = (float*)d_ws;
  float* bufA = F;                       // 16,777,216
  float* bufB = bufA + 16777216;         //  8,388,608 each
  float* bufC = bufB + 8388608;
  float* bufD = bufC + 8388608;
  float* dmat = bufD + 8388608;          // 33,554,432
  float* Pmat = dmat + 33554432;         // 33,554,432
  float* zn   = Pmat + 33554432;
  float* cn   = zn + 32768;
  float* Qv   = cn + 1024;
  float* lQ   = Qv + 1024;
  float* Qs   = lQ + 1024;
  float* scal = Qs + 1024;
  int*   idxb = (int*)(scal + 16);
  _Float16* wp = (_Float16*)(idxb + 32768);

  auto packc = [&](const float* w, int Cout, int Cin, int KH, int KW) {
    int tot = Cin * KH * KW * Cout;
    pack_conv<<<(tot + 255) / 256, 256, 0, stream>>>(w, wp, Cout, Cin, KH, KW);
  };
  auto packt = [&](const float* w, int Cout, int Cin, int KH, int KW) {
    int tot = Cin * KH * KW * Cout;
    pack_convT<<<(tot + 255) / 256, 256, 0, stream>>>(w, wp, Cout, Cin, KH, KW);
  };
  auto mkp = [&](const float* X, const float* bias, const float* resid, float* Y,
                 int Cin, int Hin, int Cout, int Hout, int KH, int stride, int pad,
                 int relu) {
    GemmP p{};
    p.X = X; p.W = wp; p.bias = bias; p.resid = resid; p.Y = Y;
    p.M = 128 * Hout * Hout; p.N = Cout; p.Ktot = Cin * KH * KH;
    p.Bn = 128; p.Cin = Cin; p.Hin = Hin; p.Win = Hin;
    p.Hout = Hout; p.Wout = Hout; p.KH = KH; p.KW = KH;
    p.stride = stride; p.pad = pad; p.relu = relu;
    return p;
  };
  auto conv = [&](const float* X, const float* bias, const float* resid, float* Y,
                  int Cin, int Hin, int Cout, int Hout, int KH, int stride, int pad,
                  int relu) {
    GemmP p = mkp(X, bias, resid, Y, Cin, Hin, Cout, Hout, KH, stride, pad, relu);
    dim3 g(p.M / 128, (p.N + 63) / 64);
    gemm_wmma<0, 0><<<g, 256, 0, stream>>>(p);
  };
  auto convT = [&](const float* X, const float* bias, float* Y,
                   int Cin, int Hin, int Cout, int Hout, int relu) {
    GemmP p = mkp(X, bias, nullptr, Y, Cin, Hin, Cout, Hout, 4, 2, 1, relu);
    dim3 g(p.M / 128, (p.N + 63) / 64);
    gemm_wmma<1, 0><<<g, 256, 0, stream>>>(p);
  };

  // ---- encoder ----
  packc(ew1, 128, 3, 4, 4);   conv(x,    eb1,  nullptr, bufA, 3,   64, 128, 32, 4, 2, 1, 1);
  packc(ew2, 256, 128, 4, 4); conv(bufA, eb2,  nullptr, bufB, 128, 32, 256, 16, 4, 2, 1, 1);
  packc(ew3, 256, 256, 3, 3); conv(bufB, eb3,  nullptr, bufC, 256, 16, 256, 16, 3, 1, 1, 1);
  packc(er1aw, 256, 256, 3, 3); conv(bufC, er1ab, nullptr, bufD, 256, 16, 256, 16, 3, 1, 1, 1);
  packc(er1bw, 256, 256, 3, 3); conv(bufD, er1bb, bufC,    bufB, 256, 16, 256, 16, 3, 1, 1, 1);
  packc(er2aw, 256, 256, 3, 3); conv(bufB, er2ab, nullptr, bufD, 256, 16, 256, 16, 3, 1, 1, 1);
  packc(er2bw, 256, 256, 3, 3); conv(bufD, er2bb, bufB,    bufC, 256, 16, 256, 16, 3, 1, 1, 1);
  // z in bufC

  // ---- distances ----
  row_norm_nchw<<<128, 256, 0, stream>>>(bufC, zn);
  cb_norm<<<4, 256, 0, stream>>>(cb, cn);
  pack_cast<<<1024, 256, 0, stream>>>(cb, wp, 262144);  // cb is already [n=code][k=d]
  {
    GemmP p{};
    p.X = bufC; p.W = wp; p.zn = zn; p.cn = cn; p.Y = dmat;
    p.M = 32768; p.N = 1024; p.Ktot = 256;
    p.Bn = 128; p.Hin = 16; p.Win = 16; p.Hout = 16; p.Wout = 16;
    gemm_wmma<2, 1><<<dim3(256, 16), 256, 0, stream>>>(p);
  }

  // ---- Blahut-Arimoto ----
  init_q<<<4, 256, 0, stream>>>(pi, Qv, lQ, Qs);
  for (int it = 0; it < 5; ++it) {
    ba_softmax<<<32768, 256, 0, stream>>>(dmat, lQ, Pmat);
    colsum<<<dim3(4, 256), 256, 0, stream>>>(Pmat, Qs);
    update_q<<<4, 256, 0, stream>>>(Qv, lQ, Qs);
  }

  // ---- zq_soft = P @ codebook (dense A via TDM) ----
  pack_cbT<<<1024, 256, 0, stream>>>(cb, wp);           // [n=d][k=code]
  {
    GemmP p{};
    p.X = Pmat; p.W = wp; p.Y = bufD;
    p.M = 32768; p.N = 256; p.Ktot = 1024;
    p.Hout = 1; p.Wout = 1;  // unused by GATHER==3/EPI==2 paths
    gemm_wmma<3, 2><<<dim3(256, 4), 256, 0, stream>>>(p);
  }
  row_argmax<<<32768, 256, 0, stream>>>(Pmat, idxb);
  scatter_zq<<<32768, 256, 0, stream>>>(cb, idxb, bufB);

  // ---- losses (before decoder overwrites z) ----
  zero_scal<<<1, 256, 0, stream>>>(scal);
  commit_reduce<<<1024, 256, 0, stream>>>(bufC, bufD, scal);

  // ---- decoder ----
  packc(dr1aw, 256, 256, 3, 3); conv(bufB, dr1ab, nullptr, bufC, 256, 16, 256, 16, 3, 1, 1, 1);
  packc(dr1bw, 256, 256, 3, 3); conv(bufC, dr1bb, bufB,    bufD, 256, 16, 256, 16, 3, 1, 1, 1);
  packc(dr2aw, 256, 256, 3, 3); conv(bufD, dr2ab, nullptr, bufB, 256, 16, 256, 16, 3, 1, 1, 1);
  packc(dr2bw, 256, 256, 3, 3); conv(bufB, dr2bb, bufD,    bufC, 256, 16, 256, 16, 3, 1, 1, 1);
  packc(dw1,   256, 256, 3, 3); conv(bufC, db1,   nullptr, bufB, 256, 16, 256, 16, 3, 1, 1, 1);
  packt(dwt2, 128, 256, 4, 4);  convT(bufB, dbt2, bufA, 256, 16, 128, 32, 1);
  packt(dwt3, 3,   128, 4, 4);  convT(bufA, dbt3, out,  128, 32, 3,   64, 0);

  finalize<<<1, 256, 0, stream>>>(scal, Qv, lQ, out + 1572864);
}
